// CrossModalAttention_34127810134238
// MI455X (gfx1250) — matrix operand order repaired and checked by reading, not compiled, not run on hardware
//
#include <hip/hip_runtime.h>
#include <hip/hip_bf16.h>

typedef unsigned short u16;
typedef unsigned int   u32;
typedef __attribute__((ext_vector_type(16))) __bf16 v16bf;
typedef __attribute__((ext_vector_type(8)))  float  v8f;

#define B_ROWS 32768
#define DMODEL 1024

__device__ __forceinline__ u16 f2bf(float f) {
    u32 u = __float_as_uint(f);
    u += 0x7fffu + ((u >> 16) & 1u);   // round to nearest even
    return (u16)(u >> 16);
}
__device__ __forceinline__ float bf2f(u16 h) {
    return __uint_as_float(((u32)h) << 16);
}
__device__ __forceinline__ v16bf pack_v16(uint4 lo, uint4 hi) {
    union { uint4 q[2]; v16bf v; } u;
    u.q[0] = lo; u.q[1] = hi;
    return u.v;
}

// ---------------- f32 -> bf16 cast (vectorized x4) ----------------
__global__ void k_cvt(const float* __restrict__ in, u16* __restrict__ out, int n4) {
    int i = blockIdx.x * blockDim.x + threadIdx.x;
    if (i >= n4) return;
    float4 f = ((const float4*)in)[i];
    ushort4 o;
    o.x = f2bf(f.x); o.y = f2bf(f.y); o.z = f2bf(f.z); o.w = f2bf(f.w);
    ((ushort4*)out)[i] = o;
}

// ---------------- epilogue helper: one 16x16 f32 fragment ----------------
// C/D layout (ISA 7.12.2): VGPR r, lanes 0-15 -> M=r, lanes 16-31 -> M=r+8; N = lane&15
template <int MODE>
__device__ __forceinline__ void store_frag(v8f c, int row, int col, int lane, int N,
                                           const float* __restrict__ bias,
                                           const float* __restrict__ resid,
                                           void* __restrict__ outp) {
    int r0 = row + ((lane & 16) ? 8 : 0);
    int n  = col + (lane & 15);
    float bs = bias[n];
#pragma unroll
    for (int r = 0; r < 8; ++r) {
        size_t idx = (size_t)(r0 + r) * N + n;
        float v = c[r] + bs;
        if (MODE == 1) {                 // + residual, keep f32 (feeds layernorm)
            v += resid[idx];
            ((float*)outp)[idx] = v;
        } else if (MODE == 2) {          // relu -> bf16
            ((u16*)outp)[idx] = f2bf(v > 0.f ? v : 0.f);
        } else {                         // bias only -> bf16
            ((u16*)outp)[idx] = f2bf(v);
        }
    }
}

// ---------------- bf16 WMMA GEMM:  out[m,n] = epi( sum_k A[m,k]*W[n,k] + bias[n] )
// A: [M,K] bf16 row-major, W: [N,K] bf16 row-major (i.e. y = A @ W^T).
// Workgroup: 256 thr = 8 waves laid out 2(M) x 4(N); WG tile 128(M) x 256(N).
// Wave tile 64x64 = 4x4 fragments -> 16 WMMAs per 16 b128 loads per K-step of 32.
// K is a compile-time constant so fragment addresses fold into load immediates.
template <int MODE, int K>
__global__ void __launch_bounds__(256)
k_gemm(const u16* __restrict__ A, const u16* __restrict__ W,
       const float* __restrict__ bias, const float* __restrict__ resid,
       void* __restrict__ outp, int N) {
    const int lane = threadIdx.x & 31;
    const int wid  = threadIdx.x >> 5;       // 0..7
    const int wm   = wid & 1;                // 0..1 (M)
    const int wn   = wid >> 1;               // 0..3 (N)
    const int row0 = blockIdx.y * 128 + wm * 64;
    const int col0 = blockIdx.x * 256 + wn * 64;

    // A fragment (16x32 bf16, ISA 7.12.2): lanes<16 hold M=lane, K = {0..7,16..23};
    // lanes>=16 hold M=lane-16, K = {8..15,24..31}.
    const int mA  = lane & 15;
    const int k0a = (lane & 16) ? 8 : 0;
    // B fragment (32x16 bf16): lanes<16 hold N=lane, K = {0..7,16..23}; lanes>=16 N=lane-16, K = {8..15,24..31}.
    const int nB  = lane & 15;
    const int k0b = (lane & 16) ? 8 : 0;

    const u16* aBase = A + (size_t)(row0 + mA) * K + k0a;
    const u16* bBase = W + (size_t)(col0 + nB) * K + k0b;

    v8f acc[4][4];
#pragma unroll
    for (int i = 0; i < 4; ++i)
#pragma unroll
        for (int j = 0; j < 4; ++j) acc[i][j] = (v8f){};

    for (int k = 0; k < K; k += 32) {
        v16bf a[4], b[4];
#pragma unroll
        for (int i = 0; i < 4; ++i) {
            const u16* p = aBase + (size_t)(16 * i) * K + k;
            a[i] = pack_v16(*(const uint4*)p, *(const uint4*)(p + 16));
        }
#pragma unroll
        for (int j = 0; j < 4; ++j) {
            const u16* p = bBase + (size_t)(16 * j) * K + k;
            b[j] = pack_v16(*(const uint4*)p, *(const uint4*)(p + 16));
        }
#pragma unroll
        for (int i = 0; i < 4; ++i)
#pragma unroll
            for (int j = 0; j < 4; ++j)
                acc[i][j] = __builtin_amdgcn_wmma_f32_16x16x32_bf16(
                    false, a[i], false, b[j], (short)0, acc[i][j], false, false);
    }

#pragma unroll
    for (int i = 0; i < 4; ++i)
#pragma unroll
        for (int j = 0; j < 4; ++j)
            store_frag<MODE>(acc[i][j], row0 + 16 * i, col0 + 16 * j,
                             lane, N, bias, resid, outp);
}

// ---------------- layernorm over D=1024: one wave per row, bf16 out ----------------
__global__ void __launch_bounds__(256)
k_ln(const float* __restrict__ in, const float* __restrict__ g,
     const float* __restrict__ bb, u16* __restrict__ out, int outStride) {
    int row  = blockIdx.x * 8 + (threadIdx.x >> 5);
    int lane = threadIdx.x & 31;
    const float* p = in + (size_t)row * DMODEL;
    float v[32];
    float s = 0.f;
#pragma unroll
    for (int j = 0; j < 32; ++j) { v[j] = p[lane + j * 32]; s += v[j]; }
#pragma unroll
    for (int off = 16; off > 0; off >>= 1) s += __shfl_xor(s, off, 32);
    float mean = s * (1.f / DMODEL);
    float q = 0.f;
#pragma unroll
    for (int j = 0; j < 32; ++j) { float d = v[j] - mean; q += d * d; }
#pragma unroll
    for (int off = 16; off > 0; off >>= 1) q += __shfl_xor(q, off, 32);
    float rstd = rsqrtf(q * (1.f / DMODEL) + 1e-5f);
    u16* o = out + (size_t)row * outStride;
#pragma unroll
    for (int j = 0; j < 32; ++j) {
        int c = lane + j * 32;
        o[c] = f2bf((v[j] - mean) * rstd * g[c] + bb[c]);
    }
}

// ---------------- final tiny matvec: out[m, 0..2] = h[m,:] @ fc2_w^T + b2 ----------------
__global__ void __launch_bounds__(256)
k_fc2(const u16* __restrict__ h, const float* __restrict__ w2,
      const float* __restrict__ b2, float* __restrict__ out) {
    int row  = blockIdx.x * 8 + (threadIdx.x >> 5);
    int lane = threadIdx.x & 31;
    const u16* hp = h + (size_t)row * DMODEL;
    float a0 = 0.f, a1 = 0.f, a2 = 0.f;
#pragma unroll 8
    for (int j = 0; j < 32; ++j) {
        int c = lane + j * 32;
        float hv = bf2f(hp[c]);
        a0 += hv * w2[c];
        a1 += hv * w2[DMODEL + c];
        a2 += hv * w2[2 * DMODEL + c];
    }
#pragma unroll
    for (int off = 16; off > 0; off >>= 1) {
        a0 += __shfl_xor(a0, off, 32);
        a1 += __shfl_xor(a1, off, 32);
        a2 += __shfl_xor(a2, off, 32);
    }
    if (lane == 0) {
        out[row * 3 + 0] = a0 + b2[0];
        out[row * 3 + 1] = a1 + b2[1];
        out[row * 3 + 2] = a2 + b2[2];
    }
}

extern "C" void kernel_launch(void* const* d_in, const int* in_sizes, int n_in,
                              void* d_out, int out_size, void* d_ws, size_t ws_size,
                              hipStream_t stream) {
    const float* text  = (const float*)d_in[0];
    const float* audio = (const float*)d_in[1];
    const float* video = (const float*)d_in[2];
    const float* w_qkv = (const float*)d_in[3];
    const float* b_qkv = (const float*)d_in[4];
    const float* w_o   = (const float*)d_in[5];
    const float* b_o   = (const float*)d_in[6];
    const float* ln_g  = (const float*)d_in[7];
    const float* ln_b  = (const float*)d_in[8];
    const float* fc1_w = (const float*)d_in[9];
    const float* fc1_b = (const float*)d_in[10];
    const float* fc2_w = (const float*)d_in[11];
    const float* fc2_b = (const float*)d_in[12];
    // d_in[13] = num_heads: unused (seq-len-1 softmax == 1, attention output == V)

    const size_t BD = (size_t)B_ROWS * DMODEL;
    const size_t DD = (size_t)DMODEL * DMODEL;

    // workspace carve-up (~660 MB total)
    char* ws = (char*)d_ws;
    u16*   xb    = (u16*)ws;   ws += 3 * BD * 2;   // bf16 text/audio/video
    u16*   wvb   = (u16*)ws;   ws += 3 * DD * 2;   // bf16 Wv[i]
    u16*   wob   = (u16*)ws;   ws += 3 * DD * 2;   // bf16 Wo[i]
    u16*   fc1b  = (u16*)ws;   ws += 3 * DD * 2;   // bf16 fc1_w [1024,3072]
    u16*   vbuf  = (u16*)ws;   ws += BD * 2;       // bf16 V
    float* tmp   = (float*)ws; ws += BD * 4;       // f32 attn+resid (LN input)
    u16*   fused = (u16*)ws;   ws += 3 * BD * 2;   // bf16 [B, 3072]
    u16*   hbuf  = (u16*)ws;   ws += BD * 2;       // bf16 relu(fc1)

    auto cvt = [&](const float* src, u16* dst, size_t n) {
        int n4 = (int)(n / 4);
        k_cvt<<<(n4 + 255) / 256, 256, 0, stream>>>(src, dst, n4);
    };
    cvt(text,  xb + 0 * BD, BD);
    cvt(audio, xb + 1 * BD, BD);
    cvt(video, xb + 2 * BD, BD);
    for (int i = 0; i < 3; ++i) {
        cvt(w_qkv + (size_t)(i * 3 + 2) * DD, wvb + i * DD, DD);  // Wv = w_qkv[i,2]
        cvt(w_o + (size_t)i * DD,             wob + i * DD, DD);
    }
    cvt(fc1_w, fc1b, 3 * DD);

    dim3 grid(DMODEL / 256, B_ROWS / 128);   // (4, 256)
    dim3 blk(256);

    // block i: xq = {text, text, audio}[i], xkv = {audio, video, video}[i]
    const int kv_idx[3] = {1, 2, 2};
    const float* xq_f32[3] = {text, text, audio};

    for (int i = 0; i < 3; ++i) {
        const float* bv = b_qkv + (size_t)(i * 3 + 2) * DMODEL;
        const float* bo = b_o + (size_t)i * DMODEL;
        // V = x_kv @ Wv^T + bv   -> bf16
        k_gemm<0, DMODEL><<<grid, blk, 0, stream>>>(xb + kv_idx[i] * BD, wvb + i * DD,
                                                    bv, nullptr, vbuf, DMODEL);
        // attn = V @ Wo^T + bo + x_q  -> f32
        k_gemm<1, DMODEL><<<grid, blk, 0, stream>>>(vbuf, wob + i * DD,
                                                    bo, xq_f32[i], tmp, DMODEL);
        // layernorm -> bf16 slice of fused [B, 3072]
        k_ln<<<B_ROWS / 8, blk, 0, stream>>>(tmp, ln_g + i * DMODEL, ln_b + i * DMODEL,
                                             fused + i * DMODEL, 3 * DMODEL);
    }
    // h = relu(fused @ fc1_w^T + b)  -> bf16
    k_gemm<2, 3 * DMODEL><<<grid, blk, 0, stream>>>(fused, fc1b, fc1_b, nullptr, hbuf,
                                                    DMODEL);
    // out = h @ fc2_w^T + b2  (B x 3, f32)
    k_fc2<<<B_ROWS / 8, blk, 0, stream>>>(hbuf, fc2_w, fc2_b, (float*)d_out);
}
